// Attention_20461224198682
// MI455X (gfx1250) — compile-verified, matrix operand order checked by reading, not run on hardware
//
#include <hip/hip_runtime.h>
#include <hip/hip_bf16.h>
#include <math.h>
#include <float.h>

typedef __attribute__((ext_vector_type(16))) __bf16 v16bf;
typedef __attribute__((ext_vector_type(8)))  float  v8f;

namespace {
constexpr int kBt   = 64;          // batch
constexpr int kN    = 196;         // tokens
constexpr int kC    = 768;         // channels
constexpr int kH    = 12;          // heads
constexpr int kD    = 64;          // head dim
constexpr int kM    = kBt * kN;    // 12544 rows (multiple of 64)
constexpr int k3C   = 3 * kC;      // 2304
constexpr int kNPad = 208;         // 13 * 16
constexpr int kWPad = 224;         // 7 * 32
constexpr int kNT   = 13;          // m-tiles per row
constexpr float kScale = 0.125f;   // 64^-0.5
}

static __device__ __forceinline__ v8f wmma_bf16(v16bf a, v16bf b, v8f c) {
  // D = A(16x32 bf16) * B(32x16 bf16) + C(f32)
  return __builtin_amdgcn_wmma_f32_16x16x32_bf16(false, a, false, b, (short)0, c,
                                                 false, false);
}

// ---------------- f32 -> bf16 conversion ----------------
__global__ void cvt_bf16_kernel(const float* __restrict__ src,
                                __bf16* __restrict__ dst, int n) {
  int i = blockIdx.x * blockDim.x + threadIdx.x;
  int stride = gridDim.x * blockDim.x;
  for (; i < n; i += stride) dst[i] = (__bf16)src[i];
}

// ---------------- dense bias gather: [H,196,196] ----------------
__global__ void bias_gather_kernel(const float* __restrict__ table,
                                   const long long* __restrict__ idx,
                                   float* __restrict__ biasBuf) {
  int i = blockIdx.x * blockDim.x + threadIdx.x;  // n*196+m
  if (i >= kN * kN) return;
  long long t = idx[i];
#pragma unroll
  for (int h = 0; h < kH; ++h)
    biasBuf[h * (kN * kN) + i] = table[t * kH + h];
}

// ---------------- zero v^T pad columns m in [196,224) ----------------
__global__ void vt_pad_kernel(__bf16* __restrict__ vt) {
  int i = blockIdx.x * blockDim.x + threadIdx.x;
  const int padw = kWPad - kN;                     // 28
  const int total = kBt * kH * kD * padw;
  if (i >= total) return;
  int m = kN + (i % padw);
  int row = i / padw;                              // (b*H+h)*D + d
  vt[(size_t)row * kWPad + m] = (__bf16)0.0f;
}

// ---- QKV GEMM: [12544,768] x [2304,768]^T, 64x32 per-wave register tile ----
__global__ void __launch_bounds__(256)
qkv_gemm_kernel(const __bf16* __restrict__ xb, const __bf16* __restrict__ wb,
                const float* __restrict__ bias, __bf16* __restrict__ qb,
                __bf16* __restrict__ kb, __bf16* __restrict__ vtb) {
  int lane = threadIdx.x & 31;
  int wave = threadIdx.x >> 5;
  int task = blockIdx.x * 8 + wave;                // 196*72 tasks, exact
  constexpr int NT32 = k3C / 32;                   // 72
  int mTile = task / NT32;
  int nTile = task % NT32;
  int m0 = mTile * 64, n0 = nTile * 32;
  int lrow = lane & 15, lhi = lane >> 4;
  const __bf16* ap = xb + (size_t)(m0 + lrow) * kC + lhi * 16;
  const __bf16* bp = wb + (size_t)(n0 + lrow) * kC + lhi * 16;
  v8f acc[4][2] = {};
  for (int kk = 0; kk < kC; kk += 32) {
    v16bf a[4], b[2];
#pragma unroll
    for (int i = 0; i < 4; ++i)
      a[i] = *(const v16bf*)(ap + (size_t)i * 16 * kC + kk);
#pragma unroll
    for (int j = 0; j < 2; ++j)
      b[j] = *(const v16bf*)(bp + (size_t)j * 16 * kC + kk);
    __builtin_prefetch(ap + kk + 128, 0, 1);
    __builtin_prefetch(bp + kk + 128, 0, 1);
#pragma unroll
    for (int i = 0; i < 4; ++i)
#pragma unroll
      for (int j = 0; j < 2; ++j)
        acc[i][j] = wmma_bf16(a[i], b[j], acc[i][j]);
  }
#pragma unroll
  for (int j = 0; j < 2; ++j) {
    int col = n0 + j * 16 + lrow;      // [0,2304)
    int sel = col / kC, c = col % kC, h = c / kD, d = c % kD;
    float bcol = bias[col];
#pragma unroll
    for (int i = 0; i < 4; ++i) {
#pragma unroll
      for (int r = 0; r < 8; ++r) {
        int t = m0 + i * 16 + lhi * 8 + r;         // token row < 12544
        int bbi = t / kN, nn = t % kN;
        float val = acc[i][j][r] + bcol;
        size_t bh = (size_t)(bbi * kH + h);
        if (sel == 0)      qb[(bh * kNPad + nn) * kD + d] = (__bf16)(val * kScale);
        else if (sel == 1) kb[(bh * kNPad + nn) * kD + d] = (__bf16)val;
        else               vtb[(bh * kD + d) * kWPad + nn] = (__bf16)val;
      }
    }
  }
}

// ---------------- attention: scores + bias + mask + softmax + P.V ----------
__global__ void __launch_bounds__(64)
attn_kernel(const __bf16* __restrict__ qb, const __bf16* __restrict__ kb,
            const __bf16* __restrict__ vtb, const float* __restrict__ biasBuf,
            const unsigned char* __restrict__ maskp, __bf16* __restrict__ ao) {
  extern __shared__ char smem[];
  constexpr int SFL = 16 * kNPad;                      // 3328 f32 scores
  constexpr int PEL = 16 * kWPad;                      // 3584 bf16 probs
  constexpr int WAVE_BYTES = SFL * 4 + PEL * 2 + 16 * 4;  // 20544
  int lane = threadIdx.x & 31;
  int wave = threadIdx.x >> 5;
  float* S = (float*)(smem + (size_t)wave * WAVE_BYTES);
  __bf16* P = (__bf16*)(smem + (size_t)wave * WAVE_BYTES + SFL * 4);
  float* rinv = (float*)(smem + (size_t)wave * WAVE_BYTES + SFL * 4 + PEL * 2);

  int task = blockIdx.x * 2 + wave;                    // 64*12*13 tasks, exact
  int bh = task / kNT, rt = task % kNT;
  int bb = bh / kH, hh = bh % kH;
  int n0 = rt * 16;
  int lrow = lane & 15, lhi = lane >> 4;

  // q fragments for this 16-row tile (D=64 -> two K=32 steps)
  const __bf16* qrow = qb + ((size_t)bh * kNPad + n0 + lrow) * kD + lhi * 16;
  v16bf qa0 = *(const v16bf*)(qrow);
  v16bf qa1 = *(const v16bf*)(qrow + 32);

  // S = q k^T (+bias, mask) for all 13 m-tiles
  for (int mt = 0; mt < kNT; ++mt) {
    int m0 = mt * 16;
    const __bf16* krow = kb + ((size_t)bh * kNPad + m0 + lrow) * kD + lhi * 16;
    v16bf kf0 = *(const v16bf*)(krow);
    v16bf kf1 = *(const v16bf*)(krow + 32);
    v8f acc = {};
    acc = wmma_bf16(qa0, kf0, acc);
    acc = wmma_bf16(qa1, kf1, acc);
    int mcol = m0 + lrow;
    bool valid = mcol < kN;
    bool keep = valid && (maskp[bb * kN + mcol] != 0);
#pragma unroll
    for (int r = 0; r < 8; ++r) {
      int nrow = lhi * 8 + r;                          // 0..15
      int ng = n0 + nrow; if (ng > kN - 1) ng = kN - 1;
      float sv;
      if (!valid)     sv = -INFINITY;                  // pad key column
      else if (!keep) sv = -FLT_MAX;                   // reference NEG_INF
      else            sv = acc[r] + biasBuf[((size_t)hh * kN + ng) * kN + mcol];
      S[nrow * kNPad + mcol] = sv;
    }
  }

  // softmax: two lanes per row, 104 columns each
  {
    float* row = S + lrow * kNPad;
    int base = lhi * 104;
    float mx = -INFINITY;
    for (int j = 0; j < 104; ++j) mx = fmaxf(mx, row[base + j]);
    mx = fmaxf(mx, __shfl_xor(mx, 16, 32));
    float sum = 0.f;
    __bf16* prow = P + lrow * kWPad;
    for (int j = 0; j < 104; ++j) {
      float e = __expf(row[base + j] - mx);            // exp(-inf)=0 on pads
      sum += e;
      prow[base + j] = (__bf16)e;
    }
    sum += __shfl_xor(sum, 16, 32);
    if (lhi == 0) rinv[lrow] = 1.0f / sum;
    for (int j = lhi * 8; j < lhi * 8 + 8; ++j)        // zero cols 208..223
      prow[kNPad + j] = (__bf16)0.0f;
  }

  // out = P @ V  (vT layout: [bh, d, m])
  const __bf16* vbase = vtb + (size_t)bh * kD * kWPad;
  for (int dt = 0; dt < 4; ++dt) {
    v8f acc = {};
    for (int kk = 0; kk < kWPad; kk += 32) {
      v16bf pa = *(const v16bf*)(P + lrow * kWPad + kk + lhi * 16);
      v16bf vf = *(const v16bf*)(vbase + (size_t)(dt * 16 + lrow) * kWPad + kk + lhi * 16);
      acc = wmma_bf16(pa, vf, acc);
    }
    int dcol = dt * 16 + lrow;
#pragma unroll
    for (int r = 0; r < 8; ++r) {
      int nrow = lhi * 8 + r;
      int ng = n0 + nrow;
      if (ng < kN)
        ao[(((size_t)bb * kN + ng) * kH + hh) * kD + dcol] =
            (__bf16)(acc[r] * rinv[nrow]);
    }
  }
}

// ---- output projection: [12544,768] x [768,768]^T + b, 64x32 per wave ------
__global__ void __launch_bounds__(256)
proj_gemm_kernel(const __bf16* __restrict__ ab, const __bf16* __restrict__ wb,
                 const float* __restrict__ bias, float* __restrict__ out) {
  int lane = threadIdx.x & 31;
  int wave = threadIdx.x >> 5;
  int task = blockIdx.x * 8 + wave;                  // 196*24 tasks, exact
  constexpr int NT32 = kC / 32;                      // 24
  int mTile = task / NT32;
  int nTile = task % NT32;
  int m0 = mTile * 64, n0 = nTile * 32;
  int lrow = lane & 15, lhi = lane >> 4;
  const __bf16* ap = ab + (size_t)(m0 + lrow) * kC + lhi * 16;
  const __bf16* bp = wb + (size_t)(n0 + lrow) * kC + lhi * 16;
  v8f acc[4][2] = {};
  for (int kk = 0; kk < kC; kk += 32) {
    v16bf a[4], b[2];
#pragma unroll
    for (int i = 0; i < 4; ++i)
      a[i] = *(const v16bf*)(ap + (size_t)i * 16 * kC + kk);
#pragma unroll
    for (int j = 0; j < 2; ++j)
      b[j] = *(const v16bf*)(bp + (size_t)j * 16 * kC + kk);
    __builtin_prefetch(ap + kk + 128, 0, 1);
    __builtin_prefetch(bp + kk + 128, 0, 1);
#pragma unroll
    for (int i = 0; i < 4; ++i)
#pragma unroll
      for (int j = 0; j < 2; ++j)
        acc[i][j] = wmma_bf16(a[i], b[j], acc[i][j]);
  }
#pragma unroll
  for (int j = 0; j < 2; ++j) {
    int col = n0 + j * 16 + lrow;
    float bc = bias[col];
#pragma unroll
    for (int i = 0; i < 4; ++i) {
#pragma unroll
      for (int r = 0; r < 8; ++r) {
        int t = m0 + i * 16 + lhi * 8 + r;
        out[(size_t)t * kC + col] = acc[i][j][r] + bc;
      }
    }
  }
}

extern "C" void kernel_launch(void* const* d_in, const int* in_sizes, int n_in,
                              void* d_out, int out_size, void* d_ws, size_t ws_size,
                              hipStream_t stream) {
  (void)in_sizes; (void)n_in; (void)out_size; (void)ws_size;
  const float*          x       = (const float*)d_in[0];
  const float*          qkv_w   = (const float*)d_in[1];
  const float*          qkv_b   = (const float*)d_in[2];
  const float*          proj_w  = (const float*)d_in[3];
  const float*          proj_b  = (const float*)d_in[4];
  const float*          rpe_tab = (const float*)d_in[5];
  const long long*      rpe_idx = (const long long*)d_in[6];
  const unsigned char*  mask    = (const unsigned char*)d_in[7];
  float*                out     = (float*)d_out;

  char* ws = (char*)d_ws;
  auto align = [](size_t v) { return (v + 255) & ~(size_t)255; };
  size_t off = 0;
  __bf16* xb      = (__bf16*)(ws + off); off = align(off + (size_t)kM * kC * 2);
  __bf16* wqkvb   = (__bf16*)(ws + off); off = align(off + (size_t)k3C * kC * 2);
  __bf16* wprojb  = (__bf16*)(ws + off); off = align(off + (size_t)kC * kC * 2);
  __bf16* qb      = (__bf16*)(ws + off); off = align(off + (size_t)kBt * kH * kNPad * kD * 2);
  __bf16* kbuf    = (__bf16*)(ws + off); off = align(off + (size_t)kBt * kH * kNPad * kD * 2);
  __bf16* vtb     = (__bf16*)(ws + off); off = align(off + (size_t)kBt * kH * kD * kWPad * 2);
  float*  biasBuf = (float*) (ws + off); off = align(off + (size_t)kH * kN * kN * 4);
  __bf16* ao      = (__bf16*)(ws + off); off = align(off + (size_t)kM * kC * 2);

  cvt_bf16_kernel<<<4096, 256, 0, stream>>>(x, xb, kM * kC);
  cvt_bf16_kernel<<<2048, 256, 0, stream>>>(qkv_w, wqkvb, k3C * kC);
  cvt_bf16_kernel<<<1024, 256, 0, stream>>>(proj_w, wprojb, kC * kC);

  bias_gather_kernel<<<(kN * kN + 255) / 256, 256, 0, stream>>>(rpe_tab, rpe_idx, biasBuf);

  {
    int total = kBt * kH * kD * (kWPad - kN);
    vt_pad_kernel<<<(total + 255) / 256, 256, 0, stream>>>(vtb);
  }

  qkv_gemm_kernel<<<(196 * 72) / 8, 256, 0, stream>>>(xb, wqkvb, qkv_b, qb, kbuf, vtb);

  {
    int tasks = kBt * kH * kNT;                 // 9984
    size_t lds = 2 * (size_t)(16 * kNPad * 4 + 16 * kWPad * 2 + 16 * 4);  // 41088
    attn_kernel<<<tasks / 2, 64, lds, stream>>>(qb, kbuf, vtb, biasBuf, mask, ao);
  }

  proj_gemm_kernel<<<(196 * 24) / 8, 256, 0, stream>>>(ao, wprojb, proj_b, out);
}
